// CausalAttention_29317446762877
// MI455X (gfx1250) — matrix operand order repaired and checked by reading, not compiled, not run on hardware
//
#include <hip/hip_runtime.h>
#include <hip/hip_bf16.h>

// ---------------------------------------------------------------------------
// Sliding-window causal attention for MI455X (gfx1250, wave32, WMMA).
// Pipeline: convert(f32->f16, transpose W) -> QKV WMMA GEMM -> flash attention
// (async K staging to LDS + WMMA scores + online softmax + WMMA PV) ->
// output-projection WMMA GEMM.
// ---------------------------------------------------------------------------

typedef __attribute__((ext_vector_type(16))) _Float16 v16h;
typedef __attribute__((ext_vector_type(8)))  _Float16 v8h;
typedef __attribute__((ext_vector_type(8)))  float    v8f;

#define B_DIM   2
#define S_LEN   2048
#define D_DIM   1024
#define NHEADS  16
#define HDIM    64
#define WINDOW  256
#define SCALE_F 0.125f      // 1/sqrt(64)
#define NTOK    (B_DIM * S_LEN)   // 4096

__device__ __forceinline__ v8f wmma_f16(v16h a, v16h b, v8f c) {
    // D = A(16x32 f16) * B(32x16 f16) + C(16x16 f32)
    return __builtin_amdgcn_wmma_f32_16x16x32_f16(
        /*neg_a=*/false, a, /*neg_b=*/false, b,
        /*c_mod=*/(short)0, c, /*reuse_a=*/false, /*reuse_b=*/false);
}

__device__ __forceinline__ v16h join8(v8h lo, v8h hi) {
    v16h r;
#pragma unroll
    for (int e = 0; e < 8; ++e) { r[e] = lo[e]; r[e + 8] = hi[e]; }
    return r;
}

__device__ __forceinline__ v8f zero8() {
    v8f z;
#pragma unroll
    for (int e = 0; e < 8; ++e) z[e] = 0.0f;
    return z;
}

// Async-copy one 32x64-f16 K chunk (4096 B) from global to LDS.
// Each of 32 lanes copies 8 x 16B; INST_OFFSET applies to both LDS and global
// addresses (ISA 10.x async addressing), so one base VGPR pair suffices.
__device__ __forceinline__ void async_copy_k_chunk(const _Float16* gbase,
                                                   unsigned int ldsbase,
                                                   int lane) {
    unsigned long long g = (unsigned long long)(size_t)gbase +
                           (unsigned long long)(lane * 16);
    unsigned int l = ldsbase + (unsigned int)(lane * 16);
    asm volatile(
        "global_load_async_to_lds_b128 %0, %1, off\n\t"
        "global_load_async_to_lds_b128 %0, %1, off offset:512\n\t"
        "global_load_async_to_lds_b128 %0, %1, off offset:1024\n\t"
        "global_load_async_to_lds_b128 %0, %1, off offset:1536\n\t"
        "global_load_async_to_lds_b128 %0, %1, off offset:2048\n\t"
        "global_load_async_to_lds_b128 %0, %1, off offset:2560\n\t"
        "global_load_async_to_lds_b128 %0, %1, off offset:3072\n\t"
        "global_load_async_to_lds_b128 %0, %1, off offset:3584"
        :: "v"(l), "v"(g) : "memory");
}

// ---------------------------------------------------------------------------
// Kernel 0: f32 -> f16 conversion; weights stored transposed WT[n][k] = W[k][n]
// ---------------------------------------------------------------------------
__global__ __launch_bounds__(256)
void swin_convert_kernel(const float* __restrict__ x,
                         const float* __restrict__ Wq, const float* __restrict__ Wk,
                         const float* __restrict__ Wv, const float* __restrict__ Wo,
                         _Float16* __restrict__ xh,
                         _Float16* __restrict__ WqT, _Float16* __restrict__ WkT,
                         _Float16* __restrict__ WvT, _Float16* __restrict__ WoT) {
    int tid = blockIdx.x * blockDim.x + threadIdx.x;
    if (tid < NTOK * D_DIM) xh[tid] = (_Float16)x[tid];
    if (tid < D_DIM * D_DIM) {
        int k = tid >> 10, n = tid & (D_DIM - 1);
        int t = n * D_DIM + k;
        WqT[t] = (_Float16)Wq[tid];
        WkT[t] = (_Float16)Wk[tid];
        WvT[t] = (_Float16)Wv[tid];
        WoT[t] = (_Float16)Wo[tid];
    }
}

// ---------------------------------------------------------------------------
// Kernel 1: QKV projection. Per wave: 32x64 output tile = 2x4 WMMA tiles.
//   A = xh [4096][1024] f16 row-major
//   B = WT [1024(n)][1024(k)] f16 (pre-transposed -> contiguous K per lane)
//   mat 0: Qh [b][h][s][64], mat 1: Kh [b][h][s][64], mat 2: VhT [b][h][hd][s]
// ---------------------------------------------------------------------------
__global__ __launch_bounds__(128)
void swin_qkv_gemm_kernel(const _Float16* __restrict__ xh,
                          const _Float16* __restrict__ WqT,
                          const _Float16* __restrict__ WkT,
                          const _Float16* __restrict__ WvT,
                          const float* __restrict__ bq,
                          const float* __restrict__ bk,
                          const float* __restrict__ bv,
                          _Float16* __restrict__ Qh,
                          _Float16* __restrict__ Kh,
                          _Float16* __restrict__ VhT) {
    const int lane   = threadIdx.x & 31;
    const int waveId = threadIdx.x >> 5;
    const int hl     = lane >> 4;        // lane half (0/1)
    const int l16    = lane & 15;
    const int rowTile = blockIdx.x * 4 + waveId;   // 0..127, 32 rows each
    const int colTile = blockIdx.y;                // 0..15,  64 cols each
    const int mat     = blockIdx.z;                // 0=Q 1=K 2=V

    const _Float16* BT   = (mat == 0) ? WqT : (mat == 1) ? WkT : WvT;
    const float*    bias = (mat == 0) ? bq  : (mat == 1) ? bk  : bv;

    v8f acc[2][4];
#pragma unroll
    for (int mi = 0; mi < 2; ++mi)
#pragma unroll
        for (int ni = 0; ni < 4; ++ni) acc[mi][ni] = zero8();

    for (int kk = 0; kk < D_DIM; kk += 32) {
        v16h a[2], b[4];
#pragma unroll
        for (int mi = 0; mi < 2; ++mi) {
            int r = rowTile * 32 + mi * 16 + l16;
            const _Float16* p = xh + r * D_DIM + kk + hl * 16;
            a[mi] = join8(*(const v8h*)p, *(const v8h*)(p + 8));
        }
#pragma unroll
        for (int ni = 0; ni < 4; ++ni) {
            int c = colTile * 64 + ni * 16 + l16;
            b[ni] = *(const v16h*)(BT + c * D_DIM + kk + hl * 16);
        }
#pragma unroll
        for (int mi = 0; mi < 2; ++mi)
#pragma unroll
            for (int ni = 0; ni < 4; ++ni)
                acc[mi][ni] = wmma_f16(a[mi], b[ni], acc[mi][ni]);
    }

#pragma unroll
    for (int mi = 0; mi < 2; ++mi) {
#pragma unroll
        for (int ni = 0; ni < 4; ++ni) {
            int n  = colTile * 64 + ni * 16 + l16;
            int h  = n >> 6, hd = n & 63;
            float bs = bias[n];
#pragma unroll
            for (int vg = 0; vg < 8; ++vg) {
                int m  = rowTile * 32 + mi * 16 + vg + hl * 8;
                int bb = m >> 11, s = m & (S_LEN - 1);
                float val = acc[mi][ni][vg] + bs;
                if (mat == 0)
                    Qh[(((bb * NHEADS) + h) * S_LEN + s) * HDIM + hd] = (_Float16)val;
                else if (mat == 1)
                    Kh[(((bb * NHEADS) + h) * S_LEN + s) * HDIM + hd] = (_Float16)val;
                else
                    VhT[(((bb * NHEADS) + h) * HDIM + hd) * S_LEN + s] = (_Float16)val;
            }
        }
    }
}

// ---------------------------------------------------------------------------
// Kernel 2: flash-style sliding-window attention. One wave = 16 query rows of
// one (b,h). 32-key chunks; double-buffered async K staging into LDS,
// WMMA scores, online softmax, WMMA PV.
// Note: j0 is 32-aligned and <= i0+15 <= 2047 => j0 <= 2016, so all K/V chunk
// accesses are in-bounds without clamping.
// ---------------------------------------------------------------------------
__global__ __launch_bounds__(128)
void swin_attn_kernel(const _Float16* __restrict__ Qh,
                      const _Float16* __restrict__ Kh,
                      const _Float16* __restrict__ VhT,
                      _Float16* __restrict__ Ah) {
    __shared__ alignas(64) _Float16 lds_p[4][16 * 40];      // P bounce, stride 40 f16
    __shared__ alignas(64) _Float16 lds_k[4][2][32 * HDIM]; // K chunks, 2 buffers/wave

    const int lane   = threadIdx.x & 31;
    const int waveId = threadIdx.x >> 5;
    const int hl     = lane >> 4;
    const int l16    = lane & 15;

    int w  = blockIdx.x * 4 + waveId;        // 0..4095
    int qt = w & 127;                        // query tile within (b,h)
    int h  = (w >> 7) & (NHEADS - 1);
    int bb = w >> 11;
    int i0 = qt * 16;

    const _Float16* Qbase = Qh  + (size_t)(bb * NHEADS + h) * S_LEN * HDIM;
    const _Float16* Kbase = Kh  + (size_t)(bb * NHEADS + h) * S_LEN * HDIM;
    const _Float16* Vbase = VhT + (size_t)(bb * NHEADS + h) * HDIM * S_LEN;
    _Float16* lp = &lds_p[waveId][0];
    // Low 32 bits of a flat pointer to LDS are the LDS byte offset.
    unsigned int kbuf_off[2] = {
        (unsigned int)(size_t)&lds_k[waveId][0][0],
        (unsigned int)(size_t)&lds_k[waveId][1][0]
    };

    // Preload Q A-fragments for hd = [0,32) and [32,64)
    v16h qa[2];
    {
        const _Float16* p0 = Qbase + (i0 + l16) * HDIM;
#pragma unroll
        for (int kh = 0; kh < 2; ++kh) {
            const _Float16* p = p0 + kh * 32 + hl * 16;
            qa[kh] = join8(*(const v8h*)p, *(const v8h*)(p + 8));
        }
    }

    v8f o[4];
#pragma unroll
    for (int ni = 0; ni < 4; ++ni) o[ni] = zero8();
    float mrow[8], lrow[8];
#pragma unroll
    for (int vg = 0; vg < 8; ++vg) { mrow[vg] = -1e30f; lrow[vg] = 0.0f; }

    int jstart = i0 - (WINDOW - 1);
    if (jstart < 0) jstart = 0;
    jstart &= ~31;

    // Kick off async copy of the first K chunk.
    async_copy_k_chunk(Kbase + (size_t)jstart * HDIM, kbuf_off[0], lane);

    int it = 0;
    for (int j0 = jstart; j0 <= i0 + 15; j0 += 32, ++it) {
        int jn = j0 + 32;
        if (jn <= i0 + 15) {
            // Prefetch next chunk into the other buffer, then retire current:
            // 16 outstanding -> wait ASYNCcnt<=8 (async loads complete in order).
            async_copy_k_chunk(Kbase + (size_t)jn * HDIM, kbuf_off[(it + 1) & 1], lane);
            asm volatile("s_wait_asynccnt 0x8" ::: "memory");
        } else {
            asm volatile("s_wait_asynccnt 0x0" ::: "memory");
        }
        const _Float16* kch = &lds_k[waveId][it & 1][0];

        // Prefetch V rows for this chunk into cache (speculative).
        __builtin_prefetch((const void*)(Vbase + j0), 0, 0);

        // ---- scores: two 16-key tiles, K fragments from LDS ----
        v8f sc[2];
#pragma unroll
        for (int kt = 0; kt < 2; ++kt) {
            v8f s = zero8();
            const _Float16* kp = kch + (kt * 16 + l16) * HDIM;
#pragma unroll
            for (int kh = 0; kh < 2; ++kh) {
                v16h bf = *(const v16h*)(kp + kh * 32 + hl * 16);
                s = wmma_f16(qa[kh], bf, s);
            }
            sc[kt] = s;
        }

        // ---- mask + scale + online softmax stats ----
        float cmax[8];
#pragma unroll
        for (int vg = 0; vg < 8; ++vg) {
            int i = i0 + vg + hl * 8;
            int d0 = i - (j0 + l16);
            float s0v = (d0 >= 0 && d0 < WINDOW) ? sc[0][vg] * SCALE_F : -1e30f;
            int d1 = i - (j0 + 16 + l16);
            float s1v = (d1 >= 0 && d1 < WINDOW) ? sc[1][vg] * SCALE_F : -1e30f;
            sc[0][vg] = s0v; sc[1][vg] = s1v;
            float t = fmaxf(s0v, s1v);
            t = fmaxf(t, __shfl_xor(t, 1));
            t = fmaxf(t, __shfl_xor(t, 2));
            t = fmaxf(t, __shfl_xor(t, 4));
            t = fmaxf(t, __shfl_xor(t, 8));
            cmax[vg] = t;
        }

#pragma unroll
        for (int vg = 0; vg < 8; ++vg) {
            float nm   = fmaxf(mrow[vg], cmax[vg]);
            float corr = __expf(mrow[vg] - nm);
            float s0v = sc[0][vg], s1v = sc[1][vg];
            float p0 = (s0v > -1e29f) ? __expf(s0v - nm) : 0.0f;
            float p1 = (s1v > -1e29f) ? __expf(s1v - nm) : 0.0f;
            sc[0][vg] = p0; sc[1][vg] = p1;
            float rs = p0 + p1;
            rs += __shfl_xor(rs, 1);
            rs += __shfl_xor(rs, 2);
            rs += __shfl_xor(rs, 4);
            rs += __shfl_xor(rs, 8);
            lrow[vg] = lrow[vg] * corr + rs;
            mrow[vg] = nm;
#pragma unroll
            for (int ni = 0; ni < 4; ++ni) o[ni][vg] *= corr;
        }

        // ---- C-layout P -> LDS (row-major 16x32, stride 40) ----
#pragma unroll
        for (int vg = 0; vg < 8; ++vg) {
            int r = vg + hl * 8;
            lp[r * 40 + l16]      = (_Float16)sc[0][vg];
            lp[r * 40 + 16 + l16] = (_Float16)sc[1][vg];
        }
        asm volatile("s_wait_dscnt 0" ::: "memory");   // wave-local LDS RAW order

        // ---- A-fragment of P from LDS ----
        v16h pa;
        {
            const _Float16* pp = lp + l16 * 40 + hl * 16;
            pa = join8(*(const v8h*)pp, *(const v8h*)(pp + 8));
        }

        // ---- PV: O(16x64) += P(16x32) x V(32x64), V transposed layout ----
#pragma unroll
        for (int ni = 0; ni < 4; ++ni) {
            int hd = ni * 16 + l16;
            v16h vb = *(const v16h*)(Vbase + hd * S_LEN + j0 + hl * 16);
            o[ni] = wmma_f16(pa, vb, o[ni]);
        }
    }

    // ---- normalize + store attention output [b][s][D] as f16 ----
#pragma unroll
    for (int vg = 0; vg < 8; ++vg) {
        float inv = 1.0f / lrow[vg];
        int s = i0 + vg + hl * 8;
#pragma unroll
        for (int ni = 0; ni < 4; ++ni) {
            int hd = ni * 16 + l16;
            Ah[((size_t)(bb * S_LEN + s)) * D_DIM + h * HDIM + hd] =
                (_Float16)(o[ni][vg] * inv);
        }
    }
}

// ---------------------------------------------------------------------------
// Kernel 3: output projection, f32 result: out = Ah @ Wo + bo
// ---------------------------------------------------------------------------
__global__ __launch_bounds__(128)
void swin_oproj_gemm_kernel(const _Float16* __restrict__ Ah,
                            const _Float16* __restrict__ WoT,
                            const float* __restrict__ bo,
                            float* __restrict__ out) {
    const int lane   = threadIdx.x & 31;
    const int waveId = threadIdx.x >> 5;
    const int hl     = lane >> 4;
    const int l16    = lane & 15;
    const int rowTile = blockIdx.x * 4 + waveId;
    const int colTile = blockIdx.y;

    v8f acc[2][4];
#pragma unroll
    for (int mi = 0; mi < 2; ++mi)
#pragma unroll
        for (int ni = 0; ni < 4; ++ni) acc[mi][ni] = zero8();

    for (int kk = 0; kk < D_DIM; kk += 32) {
        v16h a[2], b[4];
#pragma unroll
        for (int mi = 0; mi < 2; ++mi) {
            int r = rowTile * 32 + mi * 16 + l16;
            const _Float16* p = Ah + r * D_DIM + kk + hl * 16;
            a[mi] = join8(*(const v8h*)p, *(const v8h*)(p + 8));
        }
#pragma unroll
        for (int ni = 0; ni < 4; ++ni) {
            int c = colTile * 64 + ni * 16 + l16;
            b[ni] = *(const v16h*)(WoT + c * D_DIM + kk + hl * 16);
        }
#pragma unroll
        for (int mi = 0; mi < 2; ++mi)
#pragma unroll
            for (int ni = 0; ni < 4; ++ni)
                acc[mi][ni] = wmma_f16(a[mi], b[ni], acc[mi][ni]);
    }

#pragma unroll
    for (int mi = 0; mi < 2; ++mi) {
#pragma unroll
        for (int ni = 0; ni < 4; ++ni) {
            int n = colTile * 64 + ni * 16 + l16;
            float bs = bo[n];
#pragma unroll
            for (int vg = 0; vg < 8; ++vg) {
                int m = rowTile * 32 + mi * 16 + vg + hl * 8;
                out[(size_t)m * D_DIM + n] = acc[mi][ni][vg] + bs;
            }
        }
    }
}

// ---------------------------------------------------------------------------
// Launch
// ---------------------------------------------------------------------------
extern "C" void kernel_launch(void* const* d_in, const int* in_sizes, int n_in,
                              void* d_out, int out_size, void* d_ws, size_t ws_size,
                              hipStream_t stream) {
    (void)in_sizes; (void)n_in; (void)out_size;

    const float* x  = (const float*)d_in[0];
    const float* Wq = (const float*)d_in[1];
    const float* bq = (const float*)d_in[2];
    const float* Wk = (const float*)d_in[3];
    const float* bk = (const float*)d_in[4];
    const float* Wv = (const float*)d_in[5];
    const float* bv = (const float*)d_in[6];
    const float* Wo = (const float*)d_in[7];
    const float* bo = (const float*)d_in[8];
    float* out = (float*)d_out;

    const size_t MB = 1024u * 1024u;
    if (ws_size < 48u * MB) return;   // need 48 MB of scratch

    char* ws = (char*)d_ws;
    _Float16* xh  = (_Float16*)(ws);              //  8 MB: x in f16
    _Float16* WqT = (_Float16*)(ws +  8 * MB);    //  2 MB
    _Float16* WkT = (_Float16*)(ws + 10 * MB);    //  2 MB
    _Float16* WvT = (_Float16*)(ws + 12 * MB);    //  2 MB
    _Float16* WoT = (_Float16*)(ws + 14 * MB);    //  2 MB
    _Float16* Qh  = (_Float16*)(ws + 16 * MB);    //  8 MB  [b][h][s][64]
    _Float16* Kh  = (_Float16*)(ws + 24 * MB);    //  8 MB  [b][h][s][64]
    _Float16* VhT = (_Float16*)(ws + 32 * MB);    //  8 MB  [b][h][64][s]
    _Float16* Ah  = (_Float16*)(ws + 40 * MB);    //  8 MB  [b][s][1024]

    // 0: convert + transpose
    swin_convert_kernel<<<(NTOK * D_DIM) / 256, 256, 0, stream>>>(
        x, Wq, Wk, Wv, Wo, xh, WqT, WkT, WvT, WoT);

    // 1: QKV projection (z selects Q/K/V)
    swin_qkv_gemm_kernel<<<dim3(32, 16, 3), 128, 0, stream>>>(
        xh, WqT, WkT, WvT, bq, bk, bv, Qh, Kh, VhT);

    // 2: sliding-window flash attention (4096 waves / 4 per block)
    swin_attn_kernel<<<1024, 128, 0, stream>>>(Qh, Kh, VhT, Ah);

    // 3: output projection to f32
    swin_oproj_gemm_kernel<<<dim3(32, 16), 128, 0, stream>>>(Ah, WoT, bo, out);
}
